// LlamaSdpaAttention_32598801776648
// MI455X (gfx1250) — compile-verified, matrix operand order checked by reading, not run on hardware
//
#include <hip/hip_runtime.h>
#include <hip/hip_bf16.h>

// ---------------------------------------------------------------------------
// Llama SDPA attention block for gfx1250 (MI455X), wave32 + WMMA bf16.
// B=1, S=2048, HID=4096, H=32, KVH=8, D=128, N_REP=4, no causal mask.
//
// Pipeline:
//   0) f32 -> bf16 conversion of X and all weights (one streaming pass)
//   1) Q/K/V projections: bf16 WMMA GEMM (128x256 block, 64x64 per wave,
//      16 WMMAs per 8 fragments per k-step)
//   2) RoPE (+1/sqrt(D) folded into Q) bf16->bf16; V transposed to (kvh,d,S)
//   3) flash attention: 16 q-rows/wave, 32 keys/step, online softmax,
//      P staged through LDS as bf16 in WMMA A-layout
//   4) output projection -> f32 d_out
// ---------------------------------------------------------------------------

#define S_LEN 2048
#define HID   4096
#define NH    32
#define NKVH  8
#define HD    128
#define KV_N  (NKVH * HD)   // 1024

typedef __bf16 bf16_t;
typedef bf16_t v16bf __attribute__((ext_vector_type(16)));
typedef float  v8f   __attribute__((ext_vector_type(8)));

#define WMMA_BF16(A, B, C) \
    __builtin_amdgcn_wmma_f32_16x16x32_bf16(false, (A), false, (B), (short)0, (C), false, false)

// ---------------------------------------------------------------------------
// Fragment helpers (layouts per CDNA5 ISA 7.12.2, wave32)
// A 16x32 (MxK), 16-bit: lane L holds row M=L%16; K base kb=(L/16)*8;
//   halves [0..7] -> K=kb..kb+7, [8..15] -> K=16+kb..16+kb+7.
// B 32x16 (KxN) mirrors A with N=L%16.
// C/D 16x16 f32: VGPR r at lane L -> M = r + 8*(L/16), N = L%16.
// ---------------------------------------------------------------------------

__device__ inline v16bf load_frag(const bf16_t* __restrict__ base, size_t ld,
                                  int r0, int k0, int lane) {
  const int r  = r0 + (lane & 15);
  const int kb = k0 + ((lane >> 4) << 3);
  const bf16_t* p = base + (size_t)r * ld + kb;
  v16bf f;
#pragma unroll
  for (int i = 0; i < 8; ++i) f[i] = p[i];
#pragma unroll
  for (int i = 0; i < 8; ++i) f[8 + i] = p[16 + i];
  return f;
}

template <typename CT>
__device__ inline void store_c(CT* __restrict__ C, size_t ldc,
                               int r0, int n0, int lane, v8f c) {
  const int n  = n0 + (lane & 15);
  const int mb = r0 + ((lane >> 4) << 3);
#pragma unroll
  for (int r = 0; r < 8; ++r) C[(size_t)(mb + r) * ldc + n] = (CT)c[r];
}

__device__ inline float rowmax16(float v) {
  v = fmaxf(v, __shfl_xor(v, 1, 32));
  v = fmaxf(v, __shfl_xor(v, 2, 32));
  v = fmaxf(v, __shfl_xor(v, 4, 32));
  v = fmaxf(v, __shfl_xor(v, 8, 32));
  return v;
}
__device__ inline float rowsum16(float v) {
  v += __shfl_xor(v, 1, 32);
  v += __shfl_xor(v, 2, 32);
  v += __shfl_xor(v, 4, 32);
  v += __shfl_xor(v, 8, 32);
  return v;
}

// ---------------------------------------------------------------------------
// f32 -> bf16 streaming conversion (n multiple of 4)
// ---------------------------------------------------------------------------
__global__ __launch_bounds__(256) void cvt_bf16(const float* __restrict__ in,
                                                bf16_t* __restrict__ out, long n4) {
  const long i = ((long)blockIdx.x * blockDim.x + threadIdx.x);
  if (i >= n4) return;
  const float4 v = ((const float4*)in)[i];
  bf16_t o[4] = {(bf16_t)v.x, (bf16_t)v.y, (bf16_t)v.z, (bf16_t)v.w};
  *(uint2*)(out + i * 4) = *(const uint2*)o;
}

// ---------------------------------------------------------------------------
// GEMM: C[M,N] = A[M,K] * W[N,K]^T   (A, W bf16 row-major; C = CT)
// Block 256 threads = 8 waves (2x4); block tile 128x256; wave tile 64x64.
// Per k-step: 8 fragments (16 x b128 loads) -> 16 WMMAs (4x A/B reuse).
// ---------------------------------------------------------------------------
template <typename CT>
__global__ __launch_bounds__(256) void gemm_bbT(const bf16_t* __restrict__ A,
                                                const bf16_t* __restrict__ W,
                                                CT* __restrict__ C,
                                                int M, int N, int K) {
  const int lane = threadIdx.x & 31;
  const int wid  = threadIdx.x >> 5;
  const int row0 = blockIdx.x * 128 + (wid >> 2) * 64;
  const int col0 = blockIdx.y * 256 + (wid & 3) * 64;

  v8f acc[4][4];
#pragma unroll
  for (int mt = 0; mt < 4; ++mt)
#pragma unroll
    for (int nt = 0; nt < 4; ++nt) acc[mt][nt] = (v8f){};

  for (int k0 = 0; k0 < K; k0 += 32) {
    if (k0 + 32 < K) {
      __builtin_prefetch(A + (size_t)(row0 + lane) * K + k0 + 32, 0, 1);
      __builtin_prefetch(A + (size_t)(row0 + 32 + lane) * K + k0 + 32, 0, 1);
      __builtin_prefetch(W + (size_t)(col0 + lane) * K + k0 + 32, 0, 1);
      __builtin_prefetch(W + (size_t)(col0 + 32 + lane) * K + k0 + 32, 0, 1);
    }
    v16bf af[4], bf[4];
#pragma unroll
    for (int mt = 0; mt < 4; ++mt)
      af[mt] = load_frag(A, (size_t)K, row0 + 16 * mt, k0, lane);
#pragma unroll
    for (int nt = 0; nt < 4; ++nt)
      bf[nt] = load_frag(W, (size_t)K, col0 + 16 * nt, k0, lane);
#pragma unroll
    for (int mt = 0; mt < 4; ++mt)
#pragma unroll
      for (int nt = 0; nt < 4; ++nt)
        acc[mt][nt] = WMMA_BF16(af[mt], bf[nt], acc[mt][nt]);
  }

#pragma unroll
  for (int mt = 0; mt < 4; ++mt)
#pragma unroll
    for (int nt = 0; nt < 4; ++nt)
      store_c<CT>(C, (size_t)N, row0 + 16 * mt, col0 + 16 * nt, lane, acc[mt][nt]);
}

// ---------------------------------------------------------------------------
// RoPE in (S, ncol) bf16 layout, output same layout; optional scale fold.
// ---------------------------------------------------------------------------
__global__ __launch_bounds__(256) void rope_bf16(const bf16_t* __restrict__ in,
                                                 bf16_t* __restrict__ out,
                                                 int ncol, float scale) {
  const long idx = (long)blockIdx.x * blockDim.x + threadIdx.x;
  const long total = (long)S_LEN * ncol;
  if (idx >= total) return;
  const int col = (int)(idx % ncol);
  const int s   = (int)(idx / ncol);
  const int d   = col & (HD - 1);
  const int dh  = d & 63;
  const float inv_freq = __expf(-(float)dh * 0.14391157f);  // ln(1e4)/64
  const float theta = (float)s * inv_freq;
  const float c = __cosf(theta), sn = __sinf(theta);
  const float v = (float)in[idx];
  const float other = (float)in[idx + (d < 64 ? 64 : -64)];
  out[idx] = (bf16_t)(scale * (v * c + (d < 64 ? -other : other) * sn));
}

// ---------------------------------------------------------------------------
// V transpose: (S, KVH*HD) bf16 -> (KVH, HD, S) bf16
// ---------------------------------------------------------------------------
__global__ __launch_bounds__(256) void v_transpose(const bf16_t* __restrict__ in,
                                                   bf16_t* __restrict__ out) {
  const long idx = (long)blockIdx.x * blockDim.x + threadIdx.x;
  const long total = (long)S_LEN * KV_N;
  if (idx >= total) return;
  const int s   = (int)(idx % S_LEN);          // s fastest -> coalesced writes
  const int col = (int)(idx / S_LEN);          // col = kvh*HD + d
  out[(size_t)col * S_LEN + s] = in[(size_t)s * KV_N + col];
}

// ---------------------------------------------------------------------------
// Flash attention. grid (S/128, NH), 256 threads = 8 waves, 16 q-rows/wave.
// Qr: (S, NH*HD) roped+scaled bf16; Kr: (S, KVH*HD) roped bf16;
// Vt: (KVH, HD, S) bf16; O: (S, NH*HD) bf16.
// ---------------------------------------------------------------------------
__global__ __launch_bounds__(256) void attn_kernel(const bf16_t* __restrict__ Qr,
                                                   const bf16_t* __restrict__ Kr,
                                                   const bf16_t* __restrict__ Vt,
                                                   bf16_t* __restrict__ O) {
  __shared__ bf16_t ldsP[8][16 * 32];

  const int lane = threadIdx.x & 31;
  const int wid  = threadIdx.x >> 5;
  const int h    = blockIdx.y;
  const int kvh  = h >> 2;
  const int q0   = blockIdx.x * 128 + wid * 16;

  const bf16_t* Qh = Qr + (size_t)h * HD;               // ld = NH*HD
  const bf16_t* Kh = Kr + (size_t)kvh * HD;             // ld = KV_N
  const bf16_t* Vh = Vt + (size_t)kvh * HD * S_LEN;     // (HD, S), ld = S

  v16bf qf[4];
#pragma unroll
  for (int kt = 0; kt < 4; ++kt)
    qf[kt] = load_frag(Qh, (size_t)(NH * HD), q0, kt * 32, lane);

  float m_i[8], l_i[8];
#pragma unroll
  for (int r = 0; r < 8; ++r) { m_i[r] = -1e30f; l_i[r] = 0.0f; }
  v8f o[8];
#pragma unroll
  for (int nt = 0; nt < 8; ++nt) o[nt] = (v8f){};

  const int half = (lane >> 4) << 3;

  for (int j = 0; j < S_LEN; j += 32) {
    if (j + 32 < S_LEN) {
      __builtin_prefetch(Kh + (size_t)(j + 32 + (lane & 15)) * KV_N, 0, 1);
      __builtin_prefetch(Vh + (size_t)(lane & 15) * S_LEN + j + 32, 0, 1);
    }
    // scores: two 16x16 tiles over keys j..j+31 (Q already scaled)
    v8f s0 = (v8f){}, s1 = (v8f){};
#pragma unroll
    for (int kt = 0; kt < 4; ++kt) {
      const v16bf b0 = load_frag(Kh, (size_t)KV_N, j,      kt * 32, lane);
      const v16bf b1 = load_frag(Kh, (size_t)KV_N, j + 16, kt * 32, lane);
      s0 = WMMA_BF16(qf[kt], b0, s0);
      s1 = WMMA_BF16(qf[kt], b1, s1);
    }

    float alpha[8];
#pragma unroll
    for (int r = 0; r < 8; ++r) {
      const float mx = rowmax16(fmaxf(s0[r], s1[r]));
      const float mn = fmaxf(m_i[r], mx);
      alpha[r] = __expf(m_i[r] - mn);
      m_i[r] = mn;
      const float p0 = __expf(s0[r] - mn);
      const float p1 = __expf(s1[r] - mn);
      l_i[r] = l_i[r] * alpha[r] + rowsum16(p0 + p1);
      const int row = r + half;
      ldsP[wid][row * 32 + (lane & 15)]      = (bf16_t)p0;
      ldsP[wid][row * 32 + 16 + (lane & 15)] = (bf16_t)p1;
    }
    // same-wave LDS RAW ordering
    asm volatile("s_wait_dscnt 0" ::: "memory");

    // P as A-fragment (16x32 bf16 in LDS, ld=32)
    const v16bf pa = load_frag(&ldsP[wid][0], (size_t)32, 0, 0, lane);

#pragma unroll
    for (int nt = 0; nt < 8; ++nt) {
      const v16bf bv = load_frag(Vh, (size_t)S_LEN, nt * 16, j, lane);
#pragma unroll
      for (int r = 0; r < 8; ++r) o[nt][r] *= alpha[r];
      o[nt] = WMMA_BF16(pa, bv, o[nt]);
    }
  }

  // normalize, write (S, NH*HD) bf16
#pragma unroll
  for (int nt = 0; nt < 8; ++nt) {
#pragma unroll
    for (int r = 0; r < 8; ++r) {
      const int row = q0 + r + half;
      const int col = h * HD + nt * 16 + (lane & 15);
      O[(size_t)row * (NH * HD) + col] = (bf16_t)(o[nt][r] / l_i[r]);
    }
  }
}

// ---------------------------------------------------------------------------
// Host launcher
// ---------------------------------------------------------------------------
extern "C" void kernel_launch(void* const* d_in, const int* in_sizes, int n_in,
                              void* d_out, int out_size, void* d_ws, size_t ws_size,
                              hipStream_t stream) {
  const float* x  = (const float*)d_in[0];  // (S, HID)
  const float* wq = (const float*)d_in[1];  // (HID, HID)
  const float* wk = (const float*)d_in[2];  // (KV_N, HID)
  const float* wv = (const float*)d_in[3];  // (KV_N, HID)
  const float* wo = (const float*)d_in[4];  // (HID, HID)

  char* ws = (char*)d_ws;
  size_t off = 0;
  auto alloc_bf = [&](size_t elems) { bf16_t* p = (bf16_t*)(ws + off); off += elems * 2; return p; };

  bf16_t* Xb  = alloc_bf((size_t)S_LEN * HID);   // 16 MB
  bf16_t* Wqb = alloc_bf((size_t)HID * HID);     // 32 MB
  bf16_t* Wkb = alloc_bf((size_t)KV_N * HID);    //  8 MB
  bf16_t* Wvb = alloc_bf((size_t)KV_N * HID);    //  8 MB
  bf16_t* Wob = alloc_bf((size_t)HID * HID);     // 32 MB
  bf16_t* Qp  = alloc_bf((size_t)S_LEN * HID);   // raw Q proj, 16 MB
  bf16_t* Kp  = alloc_bf((size_t)S_LEN * KV_N);  //  4 MB
  bf16_t* Vp  = alloc_bf((size_t)S_LEN * KV_N);  //  4 MB
  bf16_t* Qr  = alloc_bf((size_t)S_LEN * HID);   // roped+scaled
  bf16_t* Kr  = alloc_bf((size_t)S_LEN * KV_N);
  bf16_t* Vt  = alloc_bf((size_t)S_LEN * KV_N);  // transposed
  bf16_t* Ab  = alloc_bf((size_t)S_LEN * HID);   // attention out

  const dim3 blk(256);
  auto cvt = [&](const float* src, bf16_t* dst, long n) {
    long n4 = n / 4;
    cvt_bf16<<<dim3((unsigned)((n4 + 255) / 256)), blk, 0, stream>>>(src, dst, n4);
  };

  // 0) convert everything to bf16
  cvt(x,  Xb,  (long)S_LEN * HID);
  cvt(wq, Wqb, (long)HID * HID);
  cvt(wk, Wkb, (long)KV_N * HID);
  cvt(wv, Wvb, (long)KV_N * HID);
  cvt(wo, Wob, (long)HID * HID);

  // 1) projections
  gemm_bbT<bf16_t><<<dim3(S_LEN / 128, HID / 256), blk, 0, stream>>>(Xb, Wqb, Qp, S_LEN, HID, HID);
  gemm_bbT<bf16_t><<<dim3(S_LEN / 128, KV_N / 256), blk, 0, stream>>>(Xb, Wkb, Kp, S_LEN, KV_N, HID);
  gemm_bbT<bf16_t><<<dim3(S_LEN / 128, KV_N / 256), blk, 0, stream>>>(Xb, Wvb, Vp, S_LEN, KV_N, HID);

  // 2) rope (Q gets 1/sqrt(HD) folded in), V transpose
  {
    const float sm_scale = 0.08838834764831845f;
    long nq = (long)S_LEN * HID, nk = (long)S_LEN * KV_N;
    rope_bf16<<<dim3((unsigned)((nq + 255) / 256)), blk, 0, stream>>>(Qp, Qr, HID, sm_scale);
    rope_bf16<<<dim3((unsigned)((nk + 255) / 256)), blk, 0, stream>>>(Kp, Kr, KV_N, 1.0f);
    v_transpose<<<dim3((unsigned)((nk + 255) / 256)), blk, 0, stream>>>(Vp, Vt);
  }

  // 3) attention
  attn_kernel<<<dim3(S_LEN / 128, NH), blk, 0, stream>>>(Qr, Kr, Vt, Ab);

  // 4) output projection -> f32 d_out
  gemm_bbT<float><<<dim3(S_LEN / 128, HID / 256), blk, 0, stream>>>(Ab, Wob, (float*)d_out, S_LEN, HID, HID);
}
